// EMAModule_4166118277546
// MI455X (gfx1250) — matrix operand order repaired and checked by reading, not compiled, not run on hardware
//
#include <hip/hip_runtime.h>
#include <hip/hip_bf16.h>

// EMAModule (EM attention) for MI455X / gfx1250, wave32 + WMMA f16->f32.
// B=16, C=512, N=4096 (64x64), K=64, 3 EM stages.
// Staging tiles use GLOBAL_LOAD_ASYNC_TO_LDS_B128 (ASYNCcnt) per CDNA5 ISA ch.10.

typedef __attribute__((ext_vector_type(16))) _Float16 v16h;
typedef __attribute__((ext_vector_type(8)))  _Float16 v8h;
typedef __attribute__((ext_vector_type(8)))  float    v8f;
typedef __attribute__((ext_vector_type(4)))  float    v4f;
typedef __attribute__((ext_vector_type(4)))  _Float16 v4h;

#define BB 16
#define CC 512
#define NN 4096
#define KK 64
#define EPSF 1e-6f

// A-matrix (16-bit, 16x32) element i -> K index, for lane-half hi (0/1).
__device__ __forceinline__ int amap(int i, int hi) {
    return (i < 8) ? (hi * 8 + i) : (8 + hi * 8 + i);
}

__device__ __forceinline__ v8f wmma_f16(v16h a, v16h b, v8f c) {
    // (neg_a, A, neg_b, B, c_mod, C, reuse_a, reuse_b)
    return __builtin_amdgcn_wmma_f32_16x16x32_f16(false, a, false, b, (short)0, c,
                                                  false, false);
}

// Async DMA: 16 bytes per lane, global -> LDS, tracked by ASYNCcnt.
__device__ __forceinline__ void async_b128_to_lds(void* lds_ptr, const void* gptr) {
    unsigned lofs = (unsigned)(unsigned long long)lds_ptr;  // LDS aperture: low 32b = offset
    unsigned long long ga = (unsigned long long)gptr;
    asm volatile("global_load_async_to_lds_b128 %0, %1, off"
                 :: "v"(lofs), "v"(ga) : "memory");
}
__device__ __forceinline__ void wait_async0() {
    asm volatile("s_wait_asynccnt 0x0" ::: "memory");
}

// Contiguous A fragment: two aligned 16B runs per lane.
__device__ __forceinline__ v16h load_a_contig(const _Float16* row, int hi) {
    v8h lo = *(const v8h*)(row + hi * 16);
    v8h hh = *(const v8h*)(row + hi * 16 + 8);
    return __builtin_shufflevector(lo, hh, 0, 1, 2, 3, 4, 5, 6, 7,
                                           8, 9, 10, 11, 12, 13, 14, 15);
}

// ---------------- feats fp32 -> f16 (x4 vectorized) ----------------
__global__ void k_cvt(const float* __restrict__ x, _Float16* __restrict__ y,
                      long long n4) {
    long long i = (long long)blockIdx.x * blockDim.x + threadIdx.x;
    long long stride = (long long)gridDim.x * blockDim.x;
    const v4f* x4 = (const v4f*)x;
    v4h* y4 = (v4h*)y;
    for (; i < n4; i += stride) {
        v4f v = x4[i];
        v4h h;
        h[0] = (_Float16)v[0]; h[1] = (_Float16)v[1];
        h[2] = (_Float16)v[2]; h[3] = (_Float16)v[3];
        y4[i] = h;
    }
}

// ---------------- mu0 = l2norm_C(bases), replicated to all batches ----------
__global__ void k_init_mu(const float* __restrict__ bases,
                          _Float16* __restrict__ mu_h) {
    int k = blockIdx.x;      // 0..63 (one block per basis column)
    int tid = threadIdx.x;   // 256
    __shared__ float red[256];
    float s = 0.f;
    for (int c = tid; c < CC; c += 256) {
        float v = bases[c * KK + k];
        s += v * v;
    }
    red[tid] = s;
    __syncthreads();
    for (int off = 128; off > 0; off >>= 1) {
        if (tid < off) red[tid] += red[tid + off];
        __syncthreads();
    }
    __shared__ float sinv;
    if (tid == 0) sinv = 1.0f / (EPSF + sqrtf(red[0]));
    __syncthreads();
    float inv = sinv;
    for (int c = tid; c < CC; c += 256) {
        _Float16 v = (_Float16)(bases[c * KK + k] * inv);
        for (int b = 0; b < BB; ++b)
            mu_h[((size_t)b * CC + c) * KK + k] = v;
    }
}

__global__ void k_zero(float* __restrict__ p, int n) {
    int i = blockIdx.x * blockDim.x + threadIdx.x;
    if (i < n) p[i] = 0.f;
}

// ---------------- attention = softmax_K(f^T mu); colsum accumulation --------
// grid (N/128, B), block 256. Wave w: 16 rows of N x all 64 K columns.
__global__ void __launch_bounds__(256)
k_attn(const _Float16* __restrict__ f_h, const _Float16* __restrict__ mu_h,
       _Float16* __restrict__ att_h, float* __restrict__ colsum) {
    __shared__ _Float16 smu[32 * KK];   // 32 C-rows x 64 K cols (4 KB)
    int b    = blockIdx.y;
    int wave = threadIdx.x >> 5;
    int lane = threadIdx.x & 31;
    int hi   = lane >> 4;
    int lm   = lane & 15;
    int n0   = (blockIdx.x * 8 + wave) * 16;
    const _Float16* f  = f_h  + (size_t)b * CC * NN;
    const _Float16* mu = mu_h + (size_t)b * CC * KK;

    v8f acc[4] = {};
    for (int c0 = 0; c0 < CC; c0 += 32) {
        // async DMA the 32x64 mu chunk into LDS: 16B per thread
        async_b128_to_lds((char*)smu + threadIdx.x * 16,
                          (const char*)(mu + (size_t)c0 * KK) + threadIdx.x * 16);
        wait_async0();
        __syncthreads();

        int row = n0 + lm;
        if (c0 + 32 < CC)
            __builtin_prefetch(&f[(size_t)(c0 + 32) * NN + row], 0, 0);
        v16h a;
#pragma unroll
        for (int i = 0; i < 16; ++i)
            a[i] = f[(size_t)(c0 + hi * 16 + i) * NN + row];   // A = f^T tile
#pragma unroll
        for (int t = 0; t < 4; ++t) {
            v16h bm;
#pragma unroll
            for (int i = 0; i < 16; ++i)
                bm[i] = smu[(hi * 16 + i) * KK + t * 16 + lm]; // B = mu tile
            acc[t] = wmma_f16(a, bm, acc[t]);
        }
        __syncthreads();
    }

    // softmax over the 64 K-columns of each row (row r lives in one 16-lane half)
#pragma unroll
    for (int j = 0; j < 8; ++j) {
        float m = fmaxf(fmaxf(acc[0][j], acc[1][j]), fmaxf(acc[2][j], acc[3][j]));
        for (int off = 1; off < 16; off <<= 1)
            m = fmaxf(m, __shfl_xor(m, off, 32));
        float s = 0.f;
#pragma unroll
        for (int t = 0; t < 4; ++t) {
            float e = __expf(acc[t][j] - m);
            acc[t][j] = e;
            s += e;
        }
        for (int off = 1; off < 16; off <<= 1)
            s += __shfl_xor(s, off, 32);
        float inv = 1.0f / s;
#pragma unroll
        for (int t = 0; t < 4; ++t) acc[t][j] *= inv;
    }

    // store f16 attention + per-column partial sums (l1 norm denominators)
    _Float16* att = att_h + (size_t)b * NN * KK;
#pragma unroll
    for (int t = 0; t < 4; ++t) {
        float cs = 0.f;
#pragma unroll
        for (int j = 0; j < 8; ++j) {
            int rown = n0 + j + 8 * hi;
            att[(size_t)rown * KK + t * 16 + lm] = (_Float16)acc[t][j];
            cs += acc[t][j];
        }
        cs += __shfl_xor(cs, 16, 32);
        if (hi == 0)
            atomicAdd(&colsum[b * KK + t * 16 + lm], cs);
    }
}

// ---------------- mu_raw = f @ (att / (eps+colsum)) -------------------------
// grid (C/128, B), block 256. Wave w: 16 rows of C x all 64 K columns.
__global__ void __launch_bounds__(256)
k_mu_update(const _Float16* __restrict__ f_h, const _Float16* __restrict__ att_h,
            const float* __restrict__ colsum, float* __restrict__ mu_f) {
    __shared__ _Float16 satt[32 * KK];
    int b    = blockIdx.y;
    int wave = threadIdx.x >> 5;
    int lane = threadIdx.x & 31;
    int hi   = lane >> 4;
    int lm   = lane & 15;
    int c0   = (blockIdx.x * 8 + wave) * 16;
    const _Float16* f   = f_h   + (size_t)b * CC * NN;
    const _Float16* att = att_h + (size_t)b * NN * KK;

    v8f acc[4] = {};
    for (int nb = 0; nb < NN; nb += 32) {
        async_b128_to_lds((char*)satt + threadIdx.x * 16,
                          (const char*)(att + (size_t)nb * KK) + threadIdx.x * 16);
        wait_async0();
        __syncthreads();

        const _Float16* frow = f + (size_t)(c0 + lm) * NN + nb;
        v16h a = load_a_contig(frow, hi);                  // A = f tile (2x b128)
#pragma unroll
        for (int t = 0; t < 4; ++t) {
            v16h bm;
#pragma unroll
            for (int i = 0; i < 16; ++i)
                bm[i] = satt[(hi * 16 + i) * KK + t * 16 + lm];
            acc[t] = wmma_f16(a, bm, acc[t]);
        }
        __syncthreads();
    }

    float* mub = mu_f + (size_t)b * CC * KK;
#pragma unroll
    for (int t = 0; t < 4; ++t) {
        float inv = 1.0f / (EPSF + colsum[b * KK + t * 16 + lm]);
#pragma unroll
        for (int j = 0; j < 8; ++j) {
            int c = c0 + j + 8 * hi;
            mub[(size_t)c * KK + t * 16 + lm] = acc[t][j] * inv;
        }
    }
}

// ---------------- mu = l2norm_C(mu_raw); also emit f16 copy -----------------
__global__ void k_mu_norm(float* __restrict__ mu_f, _Float16* __restrict__ mu_h) {
    int b   = blockIdx.x;
    int tid = threadIdx.x;  // 256
    int k   = tid & 63;
    int cg  = tid >> 6;     // 4 partial-sum groups per column
    float*    mub = mu_f + (size_t)b * CC * KK;
    _Float16* muh = mu_h + (size_t)b * CC * KK;

    float s = 0.f;
    for (int c = cg; c < CC; c += 4) {
        float v = mub[(size_t)c * KK + k];
        s += v * v;
    }
    __shared__ float red[256];
    __shared__ float sinv[64];
    red[tid] = s;
    __syncthreads();
    if (cg == 0) {
        float tot = red[k] + red[64 + k] + red[128 + k] + red[192 + k];
        sinv[k] = 1.0f / (EPSF + sqrtf(tot));
    }
    __syncthreads();
    for (int idx = tid; idx < CC * KK; idx += 256) {
        float v = mub[idx] * sinv[idx & 63];
        mub[idx] = v;
        muh[idx] = (_Float16)v;
    }
}

// ---------------- recon = mu @ att^T ----------------------------------------
// grid (C/128, N/64, B), block 256. Wave w: 16 rows of C x 64 cols of N.
__global__ void __launch_bounds__(256)
k_recon(const _Float16* __restrict__ mu_h, const _Float16* __restrict__ att_h,
        float* __restrict__ out) {
    int b     = blockIdx.z;
    int wave  = threadIdx.x >> 5;
    int lane  = threadIdx.x & 31;
    int hi    = lane >> 4;
    int lm    = lane & 15;
    int c0    = (blockIdx.x * 8 + wave) * 16;
    int ncol0 = blockIdx.y * 64;
    const _Float16* mu  = mu_h  + (size_t)b * CC * KK;
    const _Float16* att = att_h + (size_t)b * NN * KK;

    v8f acc[4] = {};
#pragma unroll
    for (int kb = 0; kb < KK; kb += 32) {
        const _Float16* mrow = mu + (size_t)(c0 + lm) * KK + kb;
        v16h a = load_a_contig(mrow, hi);                  // A = mu tile (2x b128)
#pragma unroll
        for (int t = 0; t < 4; ++t) {
            const _Float16* arow =
                att + (size_t)(ncol0 + t * 16 + lm) * KK + kb + hi * 16;
            v16h bm = *(const v16h*)arow;                  // B = att^T tile (32B)
            acc[t] = wmma_f16(a, bm, acc[t]);
        }
    }

    float* ob = out + (size_t)b * CC * NN;
#pragma unroll
    for (int t = 0; t < 4; ++t)
#pragma unroll
        for (int j = 0; j < 8; ++j) {
            int c = c0 + j + 8 * hi;
            ob[(size_t)c * NN + ncol0 + t * 16 + lm] = acc[t][j];
        }
}

extern "C" void kernel_launch(void* const* d_in, const int* in_sizes, int n_in,
                              void* d_out, int out_size, void* d_ws, size_t ws_size,
                              hipStream_t stream) {
    (void)in_sizes; (void)n_in; (void)out_size; (void)ws_size;
    const float* feats = (const float*)d_in[0];   // [B, C, H, W] fp32
    const float* bases = (const float*)d_in[1];   // [1, C, K]  fp32

    char* ws = (char*)d_ws;                       // all offsets 256-B aligned
    _Float16* f_h    = (_Float16*)(ws);                                      // 67108864 B
    _Float16* att_h  = (_Float16*)(ws + 67108864);                           //  8388608 B
    _Float16* mu_h   = (_Float16*)(ws + 67108864 + 8388608);                 //  1048576 B
    float*    mu_f   = (float*)(ws + 67108864 + 8388608 + 1048576);          //  2097152 B
    float*    colsum = (float*)(ws + 67108864 + 8388608 + 1048576 + 2097152);//     4096 B

    long long total4 = (long long)BB * CC * NN / 4;
    k_cvt<<<4096, 256, 0, stream>>>(feats, f_h, total4);
    k_init_mu<<<64, 256, 0, stream>>>(bases, mu_h);

    for (int s = 0; s < 3; ++s) {
        k_zero<<<4, 256, 0, stream>>>(colsum, BB * KK);
        k_attn<<<dim3(NN / 128, BB), 256, 0, stream>>>(f_h, mu_h, att_h, colsum);
        k_mu_update<<<dim3(CC / 128, BB), 256, 0, stream>>>(f_h, att_h, colsum, mu_f);
        k_mu_norm<<<BB, 256, 0, stream>>>(mu_f, mu_h);
    }
    k_recon<<<dim3(CC / 128, NN / 64, BB), 256, 0, stream>>>(mu_h, att_h,
                                                             (float*)d_out);
}
